// GatedRecurrentUnitScratch_44908178047583
// MI455X (gfx1250) — compile-verified, matrix operand order checked by reading, not run on hardware
//
#include <hip/hip_runtime.h>
#include <hip/hip_bf16.h>

// ---------------------------------------------------------------------------
// GRU-scratch pipeline for MI455X (gfx1250, wave32, WMMA + async LDS staging)
//   Stage 0: fp32 -> f16 conversion of x, Wx_{r,c,u} (concat), Wy
//   Stage 1: WMMA GEMM  XP[3][T][H] = x @ Wcat^T + bias
//            (B staged via global_load_async_to_lds_b128, double-buffered)
//   Stage 2: sequential scan (VALU matvecs, device-wide barrier per phase)
//   Stage 3: WMMA GEMM  y[T][O] = h_hist @ Wy^T + by   (same staging)
// ---------------------------------------------------------------------------

#define T_DIM 4096
#define I_DIM 1024
#define H_DIM 2048
#define O_DIM 512
#define NBLK_SCAN 64
#define BN 64        // block N-tile (cols), staged in LDS
#define BK 32        // K-chunk per WMMA step

typedef __attribute__((ext_vector_type(16))) _Float16 v16h;
typedef __attribute__((ext_vector_type(8)))  _Float16 v8h;
typedef __attribute__((ext_vector_type(8)))  float    v8f;

union Frag16 { v16h v; v8h h[2]; };

// Load one 16x32 f16 WMMA operand fragment for this lane (ISA 7.12.2):
// lane l (half=l>>4) holds K = {half*8..+7} and {16+half*8..+7}.
__device__ inline v16h load_frag(const _Float16* rowptr, int half) {
    Frag16 f;
    f.h[0] = *(const v8h*)(rowptr + half * 8);
    f.h[1] = *(const v8h*)(rowptr + 16 + half * 8);
    return f.v;
}

__device__ inline void async_copy_b128(unsigned lds_byte_off, const _Float16* gptr) {
    // Per-lane 16B global -> LDS copy, tracked by ASYNCcnt (CDNA5).
    asm volatile("global_load_async_to_lds_b128 %0, %1, off"
                 :: "v"(lds_byte_off), "v"((unsigned long long)gptr)
                 : "memory");
}

// ---------------------------------------------------------------------------
// Block GEMM core: block = 256 threads (8 waves), computes 128(M) x 64(N),
// A[MxK] row-major from global, B[NxK] row-major staged into LDS async.
// Each wave: 16 M-rows x 64 N-cols = 4 accumulators, K-loop in steps of 32.
// ---------------------------------------------------------------------------
template<int KDIM>
__device__ inline void gemm_core(const _Float16* __restrict__ A,
                                 const _Float16* __restrict__ B,
                                 int mrow0, int col0,
                                 _Float16 (*Bs)[BN * BK],   // [2][64*32] halfs
                                 v8f acc[4]) {
    const int tid  = threadIdx.x;
    const int lane = tid & 31;
    const int half = lane >> 4;
    const int idx  = lane & 15;

    // Async staging assignment: 256 lanes cover 64 rows x 4 chunks of 16B...
    // row = tid>>2 (0..63), chunk = tid&3 (0..3): 64 rows x 64B = 4KB / chunk.
    const int srow   = tid >> 2;
    const int schunk = tid & 3;
    const _Float16* bsrc = B + (size_t)(col0 + srow) * KDIM + schunk * 8;

    // Prologue: stage chunk 0 into buffer 0.
    async_copy_b128((unsigned)(unsigned long long)&Bs[0][srow * BK + schunk * 8],
                    bsrc);

    const _Float16* arow = A + (size_t)(mrow0 + idx) * KDIM;
    const int NK = KDIM / BK;
    for (int ki = 0; ki < NK; ++ki) {
        const int cur = ki & 1;
        asm volatile("s_wait_asynccnt 0x0" ::: "memory");   // my 16B copies done
        __syncthreads();                                    // chunk visible block-wide
        if (ki + 1 < NK) {                                  // overlap next chunk
            async_copy_b128(
                (unsigned)(unsigned long long)&Bs[cur ^ 1][srow * BK + schunk * 8],
                bsrc + (size_t)(ki + 1) * BK);
        }
        __builtin_prefetch(arow + ki * BK + 256, 0, 0);     // global_prefetch_b8
        v16h a = load_frag(arow + ki * BK, half);
        const _Float16* bb = &Bs[cur][0];
#pragma unroll
        for (int ns = 0; ns < 4; ++ns) {
            v16h b = load_frag(bb + (ns * 16 + idx) * BK, half);  // ds_load_b128
            acc[ns] = __builtin_amdgcn_wmma_f32_16x16x32_f16(
                false, a, false, b, (short)0, acc[ns], false, false);
        }
    }
}

// ---------------------------------------------------------------------------
// Stage 0: fp32 -> f16 conversion (grid-stride)
// ---------------------------------------------------------------------------
__global__ void cvt_f32_f16_kernel(const float* __restrict__ s,
                                   _Float16* __restrict__ d, int n) {
    int i = blockIdx.x * blockDim.x + threadIdx.x;
    int stride = gridDim.x * blockDim.x;
    for (; i < n; i += stride) d[i] = (_Float16)s[i];
}

// ---------------------------------------------------------------------------
// Init: zero h_cur / rh / z buffers and barrier state
// ---------------------------------------------------------------------------
__global__ void init_state_kernel(float* hcur, float* rh, float* zbuf,
                                  unsigned* bar) {
    for (int i = threadIdx.x; i < H_DIM; i += blockDim.x) {
        hcur[i] = 0.0f; rh[i] = 0.0f; zbuf[i] = 0.0f;
    }
    if (threadIdx.x == 0) { bar[0] = 0u; bar[1] = 0u; }
}

// ---------------------------------------------------------------------------
// Stage 1: XP[mat][T][H] = x @ [Wx_r;Wx_c;Wx_u]^T + bias
// Grid: (T/128) * (3H/64) blocks of 256 threads.
// ---------------------------------------------------------------------------
__global__ void gemm_xproj_kernel(const _Float16* __restrict__ A,   // [T][I]
                                  const _Float16* __restrict__ B,   // [3H][I]
                                  _Float16* __restrict__ XP,        // [3][T][H]
                                  const float* __restrict__ bR,
                                  const float* __restrict__ bC,
                                  const float* __restrict__ bU) {
    __shared__ alignas(16) _Float16 Bs[2][BN * BK];
    const int lane = threadIdx.x & 31;
    const int wave = threadIdx.x >> 5;
    const int nblocks = (3 * H_DIM) / BN;               // 96
    const int mblk = blockIdx.x / nblocks;
    const int nblk = blockIdx.x % nblocks;
    const int mrow0 = mblk * 128 + wave * 16;
    const int col0  = nblk * BN;

    v8f acc[4] = {};
    gemm_core<I_DIM>(A, B, mrow0, col0, Bs, acc);

    const int half = lane >> 4, idx = lane & 15;
#pragma unroll
    for (int ns = 0; ns < 4; ++ns) {
        const int col = col0 + ns * 16 + idx;
        const int mat = col >> 11;                      // / 2048
        const int hc  = col & (H_DIM - 1);
        const float bias = (mat == 0) ? bR[hc] : ((mat == 1) ? bC[hc] : bU[hc]);
        _Float16* base = XP + (size_t)mat * T_DIM * H_DIM + hc;
#pragma unroll
        for (int r = 0; r < 8; ++r) {
            int row = mrow0 + half * 8 + r;
            base[(size_t)row * H_DIM] = (_Float16)(acc[ns][r] + bias);
        }
    }
}

// ---------------------------------------------------------------------------
// Stage 2: sequential GRU scan. 64 blocks x 256 threads, device barrier.
// ---------------------------------------------------------------------------
__device__ inline void grid_barrier(volatile unsigned* bar, unsigned nblk) {
    __threadfence();
    __syncthreads();
    if (threadIdx.x == 0) {
        unsigned gen = bar[1];
        unsigned arrived = atomicAdd((unsigned*)&bar[0], 1u);
        if (arrived == nblk - 1u) {
            bar[0] = 0u;
            __threadfence();
            atomicAdd((unsigned*)&bar[1], 1u);
        } else {
            while (bar[1] == gen) { __builtin_amdgcn_s_sleep(2); }
        }
    }
    __syncthreads();
}

__global__ void scan_kernel(const _Float16* __restrict__ XP,   // [3][T][H]
                            const float* __restrict__ WhR,     // [H][H]
                            const float* __restrict__ WhC,
                            const float* __restrict__ WhU,
                            float* __restrict__ hcur,          // [H]
                            float* __restrict__ rh,            // [H]
                            float* __restrict__ zbuf,          // [H]
                            _Float16* __restrict__ hhist,      // [T][H]
                            unsigned* bar) {
    const int lane = threadIdx.x & 31;
    const int wave = threadIdx.x >> 5;          // 0..7
    const int blk  = blockIdx.x;                // 0..63
    const _Float16* XR = XP;
    const _Float16* XC = XP + (size_t)T_DIM * H_DIM;
    const _Float16* XZ = XP + 2 * (size_t)T_DIM * H_DIM;

    for (int t = 0; t < T_DIM; ++t) {
        // ---- phase 1: r = sigmoid(xr + WhR@h), z = sigmoid(xz + WhU@h) ----
        for (int rr = 0; rr < 4; ++rr) {
            const int i = blk * 32 + wave * 4 + rr;
            const float* wr = WhR + (size_t)i * H_DIM;
            const float* wu = WhU + (size_t)i * H_DIM;
            float dR = 0.0f, dU = 0.0f;
            for (int k = lane; k < H_DIM; k += 32) {
                float hk = hcur[k];
                dR += wr[k] * hk;
                dU += wu[k] * hk;
            }
#pragma unroll
            for (int off = 16; off; off >>= 1) {
                dR += __shfl_xor(dR, off, 32);
                dU += __shfl_xor(dU, off, 32);
            }
            if (lane == 0) {
                float r = 1.0f / (1.0f + expf(-((float)XR[(size_t)t * H_DIM + i] + dR)));
                float z = 1.0f / (1.0f + expf(-((float)XZ[(size_t)t * H_DIM + i] + dU)));
                rh[i]   = r * hcur[i];
                zbuf[i] = z;
            }
        }
        grid_barrier(bar, NBLK_SCAN);
        // ---- phase 2: c = tanh(xc + WhC@(r*h)); h = z*h*(1-z)*c ----------
        for (int rr = 0; rr < 4; ++rr) {
            const int i = blk * 32 + wave * 4 + rr;
            const float* wc = WhC + (size_t)i * H_DIM;
            float dC = 0.0f;
            for (int k = lane; k < H_DIM; k += 32) dC += wc[k] * rh[k];
#pragma unroll
            for (int off = 16; off; off >>= 1) dC += __shfl_xor(dC, off, 32);
            if (lane == 0) {
                float c  = tanhf((float)XC[(size_t)t * H_DIM + i] + dC);
                float z  = zbuf[i];
                float hn = z * hcur[i] * (1.0f - z) * c;   // faithful to source
                hcur[i] = hn;
                hhist[(size_t)t * H_DIM + i] = (_Float16)hn;
            }
        }
        grid_barrier(bar, NBLK_SCAN);
    }
}

// ---------------------------------------------------------------------------
// Stage 3: y[T][O] = h_hist @ Wy^T + by   (fp32 out)
// Grid: (T/128) * (O/64) blocks of 256 threads.
// ---------------------------------------------------------------------------
__global__ void gemm_y_kernel(const _Float16* __restrict__ A,   // [T][H]
                              const _Float16* __restrict__ B,   // [O][H]
                              float* __restrict__ Y,            // [T][O]
                              const float* __restrict__ by) {
    __shared__ alignas(16) _Float16 Bs[2][BN * BK];
    const int lane = threadIdx.x & 31;
    const int wave = threadIdx.x >> 5;
    const int nblocks = O_DIM / BN;                     // 8
    const int mblk = blockIdx.x / nblocks;
    const int nblk = blockIdx.x % nblocks;
    const int mrow0 = mblk * 128 + wave * 16;
    const int col0  = nblk * BN;

    v8f acc[4] = {};
    gemm_core<H_DIM>(A, B, mrow0, col0, Bs, acc);

    const int half = lane >> 4, idx = lane & 15;
#pragma unroll
    for (int ns = 0; ns < 4; ++ns) {
        const int col = col0 + ns * 16 + idx;
        const float bias = by[col];
#pragma unroll
        for (int r = 0; r < 8; ++r) {
            int row = mrow0 + half * 8 + r;
            Y[(size_t)row * O_DIM + col] = acc[ns][r] + bias;
        }
    }
}

// ---------------------------------------------------------------------------
// Host-side orchestration
// ---------------------------------------------------------------------------
extern "C" void kernel_launch(void* const* d_in, const int* in_sizes, int n_in,
                              void* d_out, int out_size, void* d_ws, size_t ws_size,
                              hipStream_t stream) {
    const float* x    = (const float*)d_in[0];
    const float* WxR  = (const float*)d_in[1];
    const float* WhR  = (const float*)d_in[2];
    const float* bR   = (const float*)d_in[3];
    const float* WxC  = (const float*)d_in[4];
    const float* WhC  = (const float*)d_in[5];
    const float* bC   = (const float*)d_in[6];
    const float* WxU  = (const float*)d_in[7];
    const float* WhU  = (const float*)d_in[8];
    const float* bU   = (const float*)d_in[9];
    const float* Wy   = (const float*)d_in[10];
    const float* by   = (const float*)d_in[11];
    float* y = (float*)d_out;

    char* ws = (char*)d_ws;
    const size_t SZ_XH    = (size_t)T_DIM * I_DIM * 2;          //  8 MB
    const size_t SZ_WCAT  = (size_t)3 * H_DIM * I_DIM * 2;      // 12 MB
    const size_t SZ_WY    = (size_t)O_DIM * H_DIM * 2;          //  2 MB
    const size_t SZ_XP    = (size_t)3 * T_DIM * H_DIM * 2;      // 48 MB
    const size_t SZ_HHIST = (size_t)T_DIM * H_DIM * 2;          // 16 MB

    _Float16* xh    = (_Float16*)(ws);
    _Float16* wcat  = (_Float16*)(ws + SZ_XH);
    _Float16* wyh   = (_Float16*)(ws + SZ_XH + SZ_WCAT);
    _Float16* xp    = (_Float16*)(ws + SZ_XH + SZ_WCAT + SZ_WY);
    _Float16* hhist = (_Float16*)(ws + SZ_XH + SZ_WCAT + SZ_WY + SZ_XP);
    char* tail      = ws + SZ_XH + SZ_WCAT + SZ_WY + SZ_XP + SZ_HHIST;
    float* hcur     = (float*)(tail);
    float* rh       = (float*)(tail + H_DIM * 4);
    float* zbuf     = (float*)(tail + 2 * H_DIM * 4);
    unsigned* bar   = (unsigned*)(tail + 3 * H_DIM * 4);

    // Stage 0: conversions + state init
    init_state_kernel<<<1, 256, 0, stream>>>(hcur, rh, zbuf, bar);
    cvt_f32_f16_kernel<<<2048, 256, 0, stream>>>(x,   xh,   T_DIM * I_DIM);
    cvt_f32_f16_kernel<<<2048, 256, 0, stream>>>(WxR, wcat,                             H_DIM * I_DIM);
    cvt_f32_f16_kernel<<<2048, 256, 0, stream>>>(WxC, wcat + (size_t)H_DIM * I_DIM,     H_DIM * I_DIM);
    cvt_f32_f16_kernel<<<2048, 256, 0, stream>>>(WxU, wcat + (size_t)2 * H_DIM * I_DIM, H_DIM * I_DIM);
    cvt_f32_f16_kernel<<<1024, 256, 0, stream>>>(Wy,  wyh,  O_DIM * H_DIM);

    // Stage 1: input projections. (T/128)=32 x (3H/64)=96 -> 3072 blocks.
    gemm_xproj_kernel<<<(T_DIM / 128) * ((3 * H_DIM) / BN), 256, 0, stream>>>(
        xh, wcat, xp, bR, bC, bU);

    // Stage 2: sequential scan with device-wide barrier
    scan_kernel<<<NBLK_SCAN, 256, 0, stream>>>(xp, WhR, WhC, WhU,
                                               hcur, rh, zbuf, hhist, bar);

    // Stage 3: output GEMM. 32 x 8 -> 256 blocks.
    gemm_y_kernel<<<(T_DIM / 128) * (O_DIM / BN), 256, 0, stream>>>(
        hhist, wyh, y, by);

    (void)in_sizes; (void)n_in; (void)out_size; (void)ws_size;
}
